// VectorQuantizer_2903397892180
// MI455X (gfx1250) — matrix operand order repaired and checked by reading, not compiled, not run on hardware
//
#include <hip/hip_runtime.h>
#include <hip/hip_bf16.h>
#include <stdint.h>

// ---------------------------------------------------------------------------
// VQ-VAE forward for MI455X (gfx1250, wave32, WMMA + async-to-LDS).
//   inputs:    (B=16, D=256, T=4096) f32
//   embedding: (K=1024, D=256) f32
//   out: [quantized_st (B,D,T) f32][vq_loss (1) f32][indices (B,T) as f32]
// Distance GEMM on v_wmma_f32_16x16x32_bf16 with hi/lo bf16 error
// compensation (3 WMMAs / tile). Codebook streamed through LDS with
// double-buffered GLOBAL_LOAD_ASYNC_TO_LDS_B128 (ASYNCcnt pipelined).
// ---------------------------------------------------------------------------

typedef __attribute__((ext_vector_type(16))) __bf16 v16bf;
typedef __attribute__((ext_vector_type(8)))  float  v8f;

union Frag { uint4 q[2]; v16bf v; };

constexpr int Bb   = 16;
constexpr int Dd   = 256;
constexpr int Tt   = 4096;
constexpr int Kk   = 1024;
constexpr int Nn   = Bb * Tt;          // 65536 rows
constexpr int QOUT = Bb * Dd * Tt;     // 16777216 floats

constexpr int ROWS   = 64;             // rows (t positions) per workgroup
constexpr int SA     = Dd + 8;         // padded LDS row: 264 halves (33 uint4)
constexpr int SA4    = SA / 8;         // uint4 per row
constexpr int CHUNK  = 32;             // codes per double-buffered chunk
constexpr int NCHUNK = Kk / CHUNK;     // 32

// LDS byte-offset layout (dynamic LDS base == 0: kernel has no static LDS).
constexpr unsigned OFF_AHI = 0;
constexpr unsigned OFF_ALO = (unsigned)(ROWS * SA * 2);        // 33792
constexpr unsigned OFF_B   = 2u * ROWS * SA * 2;               // 67584
constexpr unsigned BUFB    = 2u * CHUNK * SA * 2;              // 33792 (hi+lo)
constexpr unsigned OFF_EN  = OFF_B + 2u * BUFB;                // 135168
constexpr unsigned SMEM_BYTES = OFF_EN + 2u * CHUNK * 4u;      // 135424

__device__ __forceinline__ unsigned short f32_bf16_rn(float f) {
  unsigned u = __float_as_uint(f);
  unsigned r = u + 0x7FFFu + ((u >> 16) & 1u);
  return (unsigned short)(r >> 16);
}
__device__ __forceinline__ float bf16_f32(unsigned short h) {
  return __uint_as_float(((unsigned)h) << 16);
}

// GLOBAL_LOAD_ASYNC_TO_LDS_B128 (GVS mode): saddr = 64-bit SGPR base,
// vaddr = per-lane 32-bit byte offset, vdst = per-lane LDS byte address.
// Tracked by ASYNCcnt; 16 B per lane.
__device__ __forceinline__ void async_b128(unsigned lds_off, const void* sbase,
                                           unsigned voff) {
  asm volatile("global_load_async_to_lds_b128 %0, %1, %2"
               :
               : "v"(lds_off), "v"(voff), "s"(sbase)
               : "memory");
}

// ---------------------------------------------------------------------------
// Kernel 1: split codebook into bf16 hi/lo, compute exact ||e||^2 per code.
// ---------------------------------------------------------------------------
__global__ __launch_bounds__(256)
void vq_prep(const float* __restrict__ emb,
             unsigned short* __restrict__ ehi,
             unsigned short* __restrict__ elo,
             float* __restrict__ enorm) {
  const int k = blockIdx.x;
  const int d = threadIdx.x;
  float e = emb[(size_t)k * Dd + d];
  unsigned short h = f32_bf16_rn(e);
  float lo = e - bf16_f32(h);
  ehi[(size_t)k * Dd + d] = h;
  elo[(size_t)k * Dd + d] = f32_bf16_rn(lo);

  __shared__ float red[256];
  red[d] = e * e;
  __syncthreads();
  for (int s = 128; s > 0; s >>= 1) {
    if (d < s) red[d] += red[d + s];
    __syncthreads();
  }
  if (d == 0) enorm[k] = red[0];
}

// ---------------------------------------------------------------------------
// Kernel 2: distance GEMM + argmin.
// 128 threads (4 waves); wave w owns rows 16w..16w+15. A tile resident in
// LDS; codebook streamed via async-to-LDS double buffering. K-tile outer /
// code-tile inner loop reuses each A fragment across both code tiles.
// ---------------------------------------------------------------------------
__global__ __launch_bounds__(128)
void vq_argmin(const float* __restrict__ in,
               const unsigned short* __restrict__ ehi,
               const unsigned short* __restrict__ elo,
               const float* __restrict__ enorm,
               int* __restrict__ idxbuf,
               float* __restrict__ idxout) {
  extern __shared__ __align__(16) char smem[];
  unsigned short* sAhi = (unsigned short*)(smem + OFF_AHI);
  unsigned short* sAlo = (unsigned short*)(smem + OFF_ALO);

  const int tid  = threadIdx.x;
  const int lane = tid & 31;
  const int w    = tid >> 5;        // wave id 0..3
  const int nn   = lane & 15;       // code column / A row slot
  const int half = lane >> 4;       // wave32 half (ISA fragment layouts)
  const unsigned lane16 = (unsigned)lane * 16u;

  const int n0 = blockIdx.x * ROWS; // flat row base (b*T + t)
  const int b  = n0 >> 12;
  const int t0 = n0 & (Tt - 1);

  // Stage A tile: coalesced along t, split f32 -> bf16 hi + lo.
  for (int i = tid; i < ROWS * Dd; i += 128) {
    int d = i >> 6;
    int r = i & (ROWS - 1);
    float x = in[(size_t)b * Dd * Tt + (size_t)d * Tt + (t0 + r)];
    unsigned short h = f32_bf16_rn(x);
    sAhi[r * SA + d] = h;
    sAlo[r * SA + d] = f32_bf16_rn(x - bf16_f32(h));
  }

  float minv[8];
  int   mini[8];
#pragma unroll
  for (int j = 0; j < 8; ++j) { minv[j] = 3.4e38f; mini[j] = 0; }

  const uint4* pAhi = (const uint4*)sAhi;
  const uint4* pAlo = (const uint4*)sAlo;

  // Issue async prefetch of codebook chunk `ch` into buffer `buf`.
  // Per wave: 16 async b128 ops (8 row-groups x hi/lo) -> ASYNCcnt += 16.
  auto prefetch = [&](int ch, int buf) {
#pragma unroll
    for (int i = 0; i < CHUNK / 4; ++i) {
      int r = 4 * i + w;                               // wave w moves row r
      unsigned ldsH = OFF_B + (unsigned)buf * BUFB + (unsigned)(r * SA * 2) + lane16;
      unsigned ldsL = ldsH + (unsigned)(CHUNK * SA * 2);
      unsigned gof  = (unsigned)((ch * CHUNK + r) * (Dd * 2)) + lane16;
      async_b128(ldsH, ehi, gof);
      async_b128(ldsL, elo, gof);
    }
    if (tid < CHUNK) {
      float* sEn = (float*)(smem + OFF_EN) + buf * CHUNK;
      sEn[tid] = enorm[ch * CHUNK + tid];
    }
  };

  prefetch(0, 0);

  for (int ch = 0; ch < NCHUNK; ++ch) {
    const int buf = ch & 1;
    if (ch + 1 < NCHUNK) {
      prefetch(ch + 1, buf ^ 1);
      // Async loads complete in order: <=16 outstanding => chunk `ch` landed.
      asm volatile("s_wait_asynccnt 16" ::: "memory");
    } else {
      asm volatile("s_wait_asynccnt 0" ::: "memory");
    }
    __syncthreads();  // chunk `ch` visible to all waves (A staged on ch==0)

    const uint4* pBhi = (const uint4*)(smem + OFF_B + (unsigned)buf * BUFB);
    const uint4* pBlo = (const uint4*)(smem + OFF_B + (unsigned)buf * BUFB +
                                       (unsigned)(CHUNK * SA * 2));
    const float* sEn  = (const float*)(smem + OFF_EN) + buf * CHUNK;

    v8f acc[CHUNK / 16];
#pragma unroll
    for (int ct = 0; ct < CHUNK / 16; ++ct)
      acc[ct] = (v8f){0.f, 0.f, 0.f, 0.f, 0.f, 0.f, 0.f, 0.f};

#pragma unroll
    for (int kt = 0; kt < Dd / 32; ++kt) {
      // A 16x32 bf16 fragment (ISA layout); loaded once, reused for both
      // code tiles.
      Frag ah, al;
      int ai = (16 * w + nn) * SA4 + 4 * kt + half;
      ah.q[0] = pAhi[ai];  ah.q[1] = pAhi[ai + 2];
      al.q[0] = pAlo[ai];  al.q[1] = pAlo[ai + 2];
#pragma unroll
      for (int ct = 0; ct < CHUNK / 16; ++ct) {
        Frag bh, bl;
        int bi = (16 * ct + nn) * SA4 + 4 * kt + half;
        bh.q[0] = pBhi[bi];  bh.q[1] = pBhi[bi + 2];
        bl.q[0] = pBlo[bi];  bl.q[1] = pBlo[bi + 2];
        // dot ~= hi*hi + hi*lo + lo*hi (f32-accurate compensated product)
        acc[ct] = __builtin_amdgcn_wmma_f32_16x16x32_bf16(
            false, ah.v, false, bh.v, (short)0, acc[ct], false, false);
        acc[ct] = __builtin_amdgcn_wmma_f32_16x16x32_bf16(
            false, ah.v, false, bl.v, (short)0, acc[ct], false, false);
        acc[ct] = __builtin_amdgcn_wmma_f32_16x16x32_bf16(
            false, al.v, false, bh.v, (short)0, acc[ct], false, false);
      }
    }

#pragma unroll
    for (int ct = 0; ct < CHUNK / 16; ++ct) {
      float en   = sEn[16 * ct + nn];
      int   code = ch * CHUNK + ct * 16 + nn;
#pragma unroll
      for (int j = 0; j < 8; ++j) {
        float dist = en - 2.0f * acc[ct][j];   // ||e||^2 - 2 x.e
        if (dist < minv[j]) { minv[j] = dist; mini[j] = code; }
      }
    }
    __syncthreads();  // all waves done reading buf before it is re-filled
  }

  // Cross-lane argmin: row m lives in 16 lanes; both wave halves reduce in
  // parallel. Tie -> lowest index, matching jnp.argmin.
#pragma unroll
  for (int j = 0; j < 8; ++j) {
    float v  = minv[j];
    int   ix = mini[j];
#pragma unroll
    for (int off = 8; off > 0; off >>= 1) {
      float ov = __shfl_xor(v, off, 32);
      int   oi = __shfl_xor(ix, off, 32);
      if (ov < v || (ov == v && oi < ix)) { v = ov; ix = oi; }
    }
    if (nn == 0) {
      int row = 16 * w + j + 8 * half;   // C layout: VGPR j -> rows j / j+8
      int n   = n0 + row;
      idxbuf[n] = ix;
      idxout[n] = (float)ix;
    }
  }
}

// ---------------------------------------------------------------------------
// Kernel 3: gather quantized rows (exact f32), write (B,D,T) output,
// accumulate per-block squared-error partials (deterministic, no atomics).
// ---------------------------------------------------------------------------
__global__ __launch_bounds__(256)
void vq_gather(const float* __restrict__ in,
               const float* __restrict__ emb,
               const int* __restrict__ idxbuf,
               float* __restrict__ out,
               float* __restrict__ partials) {
  const int n = blockIdx.x * 256 + threadIdx.x;
  const int b = n >> 12;
  const int t = n & (Tt - 1);
  const int ix = idxbuf[n];
  const float* e = emb + (size_t)ix * Dd;
  const size_t base = (size_t)b * Dd * Tt + t;
  float s = 0.f;
#pragma unroll 4
  for (int d = 0; d < Dd; ++d) {
    float q = e[d];                          // L2-resident codebook
    float x = in[base + (size_t)d * Tt];     // coalesced
    out[base + (size_t)d * Tt] = q;          // straight-through forward == q
    float df = q - x;
    s += df * df;
  }
  __shared__ float red[256];
  red[threadIdx.x] = s;
  __syncthreads();
  for (int st = 128; st > 0; st >>= 1) {
    if (threadIdx.x < st) red[threadIdx.x] += red[threadIdx.x + st];
    __syncthreads();
  }
  if (threadIdx.x == 0) partials[blockIdx.x] = red[0];
}

// ---------------------------------------------------------------------------
// Kernel 4: vq_loss = (1 + commitment_cost) * mean squared error.
// ---------------------------------------------------------------------------
__global__ __launch_bounds__(256)
void vq_loss_finalize(const float* __restrict__ partials,
                      float* __restrict__ lossOut) {
  __shared__ float red[256];
  red[threadIdx.x] = partials[threadIdx.x];
  __syncthreads();
  for (int st = 128; st > 0; st >>= 1) {
    if (threadIdx.x < st) red[threadIdx.x] += red[threadIdx.x + st];
    __syncthreads();
  }
  if (threadIdx.x == 0)
    lossOut[0] = 1.25f * red[0] / (float)((size_t)Nn * Dd);
}

// ---------------------------------------------------------------------------
extern "C" void kernel_launch(void* const* d_in, const int* in_sizes, int n_in,
                              void* d_out, int out_size, void* d_ws, size_t ws_size,
                              hipStream_t stream) {
  (void)in_sizes; (void)n_in; (void)out_size; (void)ws_size;
  const float* in  = (const float*)d_in[0];   // (B, D, T) f32
  const float* emb = (const float*)d_in[1];   // (K, D) f32
  float* out = (float*)d_out;

  // Workspace carve (~1.26 MB).
  unsigned short* ehi   = (unsigned short*)d_ws;                 // K*D bf16
  unsigned short* elo   = ehi + (size_t)Kk * Dd;                 // K*D bf16
  float*          enorm = (float*)(elo + (size_t)Kk * Dd);       // K f32
  int*            idxb  = (int*)(enorm + Kk);                    // N i32
  float*          part  = (float*)(idxb + Nn);                   // Nn/256 f32

  float* lossOut = out + QOUT;
  float* idxOut  = out + QOUT + 1;

  vq_prep<<<Kk, 256, 0, stream>>>(emb, ehi, elo, enorm);

  vq_argmin<<<Nn / ROWS, 128, SMEM_BYTES, stream>>>(in, ehi, elo, enorm,
                                                    idxb, idxOut);

  vq_gather<<<Nn / 256, 256, 0, stream>>>(in, emb, idxb, out, part);
  vq_loss_finalize<<<1, 256, 0, stream>>>(part, lossOut);
}
